// DenseCLIP_56556129354110
// MI455X (gfx1250) — matrix operand order repaired and checked by reading, not compiled, not run on hardware
//
#include <hip/hip_runtime.h>
#include <hip/hip_bf16.h>
#include <float.h>

typedef __attribute__((ext_vector_type(16))) _Float16 v16h;
typedef __attribute__((ext_vector_type(8)))  _Float16 v8h;
typedef __attribute__((ext_vector_type(8)))  float    v8f;

constexpr int kB  = 64;    // batch (both text & video)
constexpr int kT1 = 33;    // text tokens incl. CLS
constexpr int kTT = 32;    // tokens used (drop token 0)
constexpr int kC  = 512;   // channels
constexpr int kI  = 197;   // video frames
constexpr int kIP = 208;   // frames padded to 13*16
constexpr int kNT = 13;    // i-tiles of 16
constexpr float kTemp = 0.07f;

// ---------- Kernel 1a: normalize text along token axis (axis=1), drop token 0, emit f16 ----------
__global__ __launch_bounds__(256)
void norm_text_kernel(const float* __restrict__ text, _Float16* __restrict__ TL) {
  int tid = blockIdx.x * blockDim.x + threadIdx.x;   // b*512 + c
  int b = tid >> 9;
  int c = tid & 511;
  const float* col = text + (size_t)b * kT1 * kC + c;
  float s = 0.0f;
  #pragma unroll
  for (int t = 0; t < kT1; ++t) { float v = col[t * kC]; s += v * v; }
  float inv = 1.0f / sqrtf(s);
  _Float16* dst = TL + (size_t)b * kTT * kC + c;
  #pragma unroll
  for (int t = 0; t < kTT; ++t)
    dst[t * kC] = (_Float16)(col[(t + 1) * kC] * inv);
}

// ---------- Kernel 1b: normalize video along frame axis, emit f16, zero pad rows ----------
__global__ __launch_bounds__(256)
void norm_video_kernel(const float* __restrict__ video, _Float16* __restrict__ VE) {
  int tid = blockIdx.x * blockDim.x + threadIdx.x;   // b*512 + c
  int b = tid >> 9;
  int c = tid & 511;
  const float* col = video + (size_t)b * kI * kC + c;
  float s = 0.0f;
  for (int i = 0; i < kI; ++i) { float v = col[i * kC]; s += v * v; }
  float inv = 1.0f / sqrtf(s);
  _Float16* dst = VE + (size_t)b * kIP * kC + c;
  for (int i = 0; i < kI; ++i) dst[i * kC] = (_Float16)(col[i * kC] * inv);
  for (int i = kI; i < kIP; ++i) dst[i * kC] = (_Float16)0.0f;
}

// ---------- Kernel 2: per (x,y): 32x197 sim GEMM via WMMA, row-max over i, masked mean over t ----------
__global__ __launch_bounds__(416)
void sim_kernel(const _Float16* __restrict__ TL, const _Float16* __restrict__ VE,
                const int* __restrict__ amask, float* __restrict__ tti) {
  __shared__ float rowmax[kTT][kNT];
  const int x    = blockIdx.x;
  const int y    = blockIdx.y;
  const int wave = threadIdx.x >> 5;   // 0..12, one i-tile per wave
  const int lane = threadIdx.x & 31;
  const int hgrp = lane >> 4;          // half-wave group
  const int r    = lane & 15;
  const int i0   = wave * 16;

  // A fragments: rows t=r and t=16+r of TL_x (row-major [t][d])
  const _Float16* a0p = TL + ((size_t)x * kTT + r) * kC;
  const _Float16* a1p = a0p + 16 * kC;
  // B fragment: row i = i0+r of VE_y (B[k][n] = VE[n][k])
  const _Float16* bp  = VE + ((size_t)y * kIP + i0 + r) * kC;

  v8f d0 = {};  // t rows 0..15  x  i cols i0..i0+15
  v8f d1 = {};  // t rows 16..31
  for (int kk = 0; kk < kC; kk += 32) {
    const int klo = kk + hgrp * 8;      // A: K[klo..klo+7] and K[klo+16..klo+23]
    const int khi = klo + 16;
    v8h a0l = *(const v8h*)(a0p + klo);
    v8h a0h = *(const v8h*)(a0p + khi);
    v8h a1l = *(const v8h*)(a1p + klo);
    v8h a1h = *(const v8h*)(a1p + khi);
    v16h bf = __builtin_shufflevector(*(const v8h*)(bp + klo), *(const v8h*)(bp + khi), 0,1,2,3,4,5,6,7,8,9,10,11,12,13,14,15);   // B: K[klo..klo+7] and K[klo+16..klo+23]
    v16h af0 = __builtin_shufflevector(a0l, a0h, 0,1,2,3,4,5,6,7,8,9,10,11,12,13,14,15);
    v16h af1 = __builtin_shufflevector(a1l, a1h, 0,1,2,3,4,5,6,7,8,9,10,11,12,13,14,15);
    d0 = __builtin_amdgcn_wmma_f32_16x16x32_f16(false, af0, false, bf, (short)0, d0, false, false);
    d1 = __builtin_amdgcn_wmma_f32_16x16x32_f16(false, af1, false, bf, (short)0, d1, false, false);
  }

  // scale by temperature, invalidate padded columns, per-row max over this tile's 16 columns.
  const bool iok = (i0 + r) < kI;      // C/D: col index == lane&15
  #pragma unroll
  for (int v = 0; v < 8; ++v) {
    float m0 = iok ? (float)d0[v] * kTemp : -FLT_MAX;
    float m1 = iok ? (float)d1[v] * kTemp : -FLT_MAX;
    #pragma unroll
    for (int off = 1; off < 16; off <<= 1) {   // xor<16 stays within the 16-lane half-group
      m0 = fmaxf(m0, __shfl_xor(m0, off));
      m1 = fmaxf(m1, __shfl_xor(m1, off));
    }
    if (r == 0) {
      rowmax[v + 8 * hgrp][wave]      = m0;   // C/D row = vgpr + 8*(lane>=16)
      rowmax[16 + v + 8 * hgrp][wave] = m1;
    }
  }
  __syncthreads();

  // wave 0: combine 13 tile maxima per token, masked mean over tokens
  if (threadIdx.x < kTT) {
    const int t = threadIdx.x;
    float m = -FLT_MAX;
    #pragma unroll
    for (int w = 0; w < kNT; ++w) m = fmaxf(m, rowmax[t][w]);
    const float mk = (amask[x * kT1 + 1 + t] != 0) ? 1.0f : 0.0f;
    float num = mk * m;
    float cnt = mk;
    #pragma unroll
    for (int off = 1; off < 32; off <<= 1) {
      num += __shfl_xor(num, off);
      cnt += __shfl_xor(cnt, off);
    }
    if (t == 0) tti[x * kB + y] = num / fmaxf(cnt, 1e-6f);
  }
}

// ---------- Kernel 3: contrastive loss over 64x64 tti ----------
__global__ __launch_bounds__(64)
void loss_kernel(const float* __restrict__ tti, float* __restrict__ out) {
  __shared__ float red[kB];
  const int x = threadIdx.x;
  float s = 0.0f;
  for (int y = 0; y < kB; ++y) s += expf(tti[x * kB + y]);
  const float pos = expf(tti[x * kB + x]);
  red[x] = -logf(pos / s + 1e-20f);
  __syncthreads();
  for (int st = 32; st > 0; st >>= 1) {
    if (x < st) red[x] += red[x + st];
    __syncthreads();
  }
  if (x == 0) out[0] = red[0] * (1.0f / (float)kB);
}

extern "C" void kernel_launch(void* const* d_in, const int* in_sizes, int n_in,
                              void* d_out, int out_size, void* d_ws, size_t ws_size,
                              hipStream_t stream) {
  (void)in_sizes; (void)n_in; (void)out_size; (void)ws_size;
  const float* text  = (const float*)d_in[0];   // (64,33,512) f32
  const float* video = (const float*)d_in[1];   // (64,197,512) f32
  const int*   amask = (const int*)d_in[2];     // (64,33) i32
  float* out = (float*)d_out;                   // (1,) f32

  char* ws = (char*)d_ws;
  const size_t tl_bytes = (size_t)kB * kTT * kC * sizeof(_Float16);  // 2 MB
  const size_t ve_bytes = (size_t)kB * kIP * kC * sizeof(_Float16);  // 13 MB
  _Float16* TL  = (_Float16*)ws;
  _Float16* VE  = (_Float16*)(ws + tl_bytes);
  float*    tti = (float*)(ws + tl_bytes + ve_bytes);                // 16 KB

  norm_text_kernel <<<(kB * kC) / 256, 256, 0, stream>>>(text, TL);
  norm_video_kernel<<<(kB * kC) / 256, 256, 0, stream>>>(video, VE);
  dim3 grid(kB, kB);
  sim_kernel<<<grid, 13 * 32, 0, stream>>>(TL, VE, amask, tti);
  loss_kernel<<<1, kB, 0, stream>>>(tti, out);
}
